// RegionProposalNetwork_41944650612768
// MI455X (gfx1250) — matrix operand order repaired and checked by reading, not compiled, not run on hardware
//
#include <hip/hip_runtime.h>
#include <cstdint>
#include <cmath>

// ---------------------------------------------------------------------------
// RPN for Faster-RCNN on MI455X (gfx1250).
// Dominant cost: 3x3 conv 512->512 over 4x64x64 = implicit GEMM
//   M=16384, N=512, K=4608 -> 77.3 GFLOP, ~76MB traffic (L2 resident).
// Compute-bound -> v_wmma_f32_16x16x32_bf16 with fp32 accumulation.
// K reordered as k' = (ky*3+kx)*512 + c so each 32-wide K tile has a
// uniform filter tap -> im2col addressing is adds only (no divisions).
// NMS: selection-sort argmax loop, scores held in LDS (320KB/WGP on CDNA5).
// ---------------------------------------------------------------------------

typedef __attribute__((ext_vector_type(16))) __bf16 v16bf;
typedef __attribute__((ext_vector_type(8)))  float  v8f;

union Frag16 {
    v16bf v;
    uint4 q[2];
};

// ---- bf16 helpers (storage type = unsigned short at ABI boundaries) -------
__device__ __forceinline__ unsigned short f32_to_bf16(float f) {
    unsigned u; __builtin_memcpy(&u, &f, 4);
    unsigned r = (u + 0x7FFFu + ((u >> 16) & 1u)) >> 16;   // round-to-nearest-even
    return (unsigned short)r;
}
__device__ __forceinline__ float bf16lo_f32(unsigned packed) {
    unsigned u = packed << 16; float f; __builtin_memcpy(&f, &u, 4); return f;
}
__device__ __forceinline__ float bf16hi_f32(unsigned packed) {
    unsigned u = packed & 0xFFFF0000u; float f; __builtin_memcpy(&f, &u, 4); return f;
}

// ---- problem constants ----------------------------------------------------
constexpr int B_    = 4;
constexpr int C_    = 512;
constexpr int H_    = 64;
constexpr int W_    = 64;
constexpr int A_    = 9;
constexpr int M_    = B_ * H_ * W_;      // 16384
constexpr int K_    = C_ * 9;            // 4608
constexpr int N_    = 512;
constexpr int HP    = 66;                // padded spatial
constexpr int HP2   = HP * HP;           // 4356
constexpr int NANCH = H_ * W_ * A_;      // 36864
constexpr int NPOST = 2000;

// workspace layout (bytes)
constexpr size_t OFF_XP  = 0;                                  // bf16 [B][C][66][66]
constexpr size_t SZ_XP   = (size_t)B_ * C_ * HP2 * 2;
constexpr size_t OFF_WT  = OFF_XP + SZ_XP;                     // bf16 [N][K'] (tap-major K)
constexpr size_t SZ_WT   = (size_t)N_ * K_ * 2;
constexpr size_t OFF_H   = OFF_WT + SZ_WT;                     // bf16 [M][512]
constexpr size_t SZ_H    = (size_t)M_ * N_ * 2;
constexpr size_t OFF_LR  = OFF_H + SZ_H;                       // f32 [M][36]
constexpr size_t SZ_LR   = (size_t)M_ * 36 * 4;
constexpr size_t OFF_SR  = OFF_LR + SZ_LR;                     // f32 [M][18]
constexpr size_t SZ_SR   = (size_t)M_ * 18 * 4;
constexpr size_t OFF_BX  = OFF_SR + SZ_SR;                     // f32 [B][36864][4]
constexpr size_t SZ_BX   = (size_t)B_ * NANCH * 4 * 4;
constexpr size_t OFF_SC  = OFF_BX + SZ_BX;                     // f32 [B][36864]

// d_out layout (f32 elements, tuple concatenated)
constexpr size_t OUT_LOCS   = 0;                               // 4*36864*4
constexpr size_t OUT_SCORES = OUT_LOCS + (size_t)B_ * NANCH * 4;
constexpr size_t OUT_ROIS   = OUT_SCORES + (size_t)B_ * NANCH * 2;
constexpr size_t OUT_INDS   = OUT_ROIS + (size_t)B_ * NPOST * 4;
constexpr size_t OUT_ANCHOR = OUT_INDS + (size_t)B_ * NPOST;

// ---------------------------------------------------------------------------
// 1) pad+convert input activations to bf16:  xp[b][c][66][66]
// ---------------------------------------------------------------------------
__global__ void prep_x_kernel(const float* __restrict__ x, unsigned short* __restrict__ xp) {
    size_t id = (size_t)blockIdx.x * blockDim.x + threadIdx.x;
    size_t total = (size_t)B_ * C_ * HP2;
    if (id >= total) return;
    int p  = (int)(id % HP2);
    int bc = (int)(id / HP2);
    int yy = p / HP, xx = p % HP;
    float v = 0.0f;
    if (yy >= 1 && yy <= H_ && xx >= 1 && xx <= W_) {
        v = x[(size_t)bc * (H_ * W_) + (yy - 1) * W_ + (xx - 1)];
    }
    xp[id] = f32_to_bf16(v);
}

// ---------------------------------------------------------------------------
// 2) conv1 weights OIHW -> bf16 tap-major GEMM layout Wt[n][t*512 + c],
//    t = ky*3+kx.  (K permutation only; GEMM reduction order-invariant.)
// ---------------------------------------------------------------------------
__global__ void prep_w_kernel(const float* __restrict__ w, unsigned short* __restrict__ Wt) {
    size_t id = (size_t)blockIdx.x * blockDim.x + threadIdx.x;
    if (id >= (size_t)N_ * K_) return;
    int o = (int)(id / K_);
    int k = (int)(id - (size_t)o * K_);
    int t = k >> 9;          // tap index 0..8
    int c = k & 511;         // input channel
    Wt[id] = f32_to_bf16(w[((size_t)o * 512 + c) * 9 + t]);
}

// ---------------------------------------------------------------------------
// 3) 3x3 conv as implicit GEMM with WMMA bf16.
//    Block tile 128(M) x 128(N), K-step 32. 8 waves: 4(M) x 2(N),
//    wave tile 32x64 -> 2x4 accumulators of v8f.
// ---------------------------------------------------------------------------
#define LDSROW 40   // 32 + 8 pad (bf16 units); 80B rows keep 16B alignment

__global__ __launch_bounds__(256)
void conv3x3_wmma_kernel(const unsigned short* __restrict__ xp,
                         const unsigned short* __restrict__ Wt,
                         const float* __restrict__ bias,
                         unsigned short* __restrict__ h) {
    __shared__ unsigned short As[128][LDSROW];
    __shared__ unsigned short Bs[128][LDSROW];

    const int tid  = threadIdx.x;
    const int lane = tid & 31;
    const int wave = tid >> 5;
    const int wm   = wave & 3;        // M wave index (0..3)
    const int wn   = wave >> 2;       // N wave index (0..1)
    const int m0   = blockIdx.x * 128;
    const int n0   = blockIdx.y * 128;

    const int half = lane >> 4;       // 0 or 1
    const int l15  = lane & 15;
    const int kA   = half * 8;        // A frag K base within 16 (ISA 7.12.2)
    const int kB   = half * 16;       // unused: B frag loads use kA, same k order as A

    // Loop-invariant im2col bases: element e -> (mm, kk) fixed per thread.
    // base_e = address of xp[b][kk][y][xx]; per K-step add uniform
    // soff = c0*4356 + ky*66 + kx.
    int baseE[16];
#pragma unroll
    for (int e = 0; e < 16; ++e) {
        int idx = tid + e * 256;
        int mm = idx & 127;
        int kk = idx >> 7;
        int m  = m0 + mm;
        int b  = m >> 12;
        int rm = m & 4095;
        int y  = rm >> 6;
        int xx = rm & 63;
        baseE[e] = (((b << 9) + kk) * HP + y) * HP + xx;
    }

    v8f acc[2][4];
    v8f z = {0.f, 0.f, 0.f, 0.f, 0.f, 0.f, 0.f, 0.f};
#pragma unroll
    for (int mt = 0; mt < 2; ++mt)
#pragma unroll
        for (int nt = 0; nt < 4; ++nt) acc[mt][nt] = z;

    for (int k0 = 0; k0 < K_; k0 += 32) {
        // uniform (scalar) per K-step: filter tap + channel base
        int t  = k0 >> 9;             // tap 0..8 (tile never straddles taps)
        int c0 = k0 & 511;
        int ky = t / 3;
        int kx = t - ky * 3;
        const unsigned short* xs = xp + (c0 * HP2 + ky * HP + kx);

        // ---- stage A tile: 128 rows (im2col) x 32 K; adds-only addressing ----
#pragma unroll
        for (int e = 0; e < 16; ++e) {
            int idx = tid + e * 256;
            As[idx & 127][idx >> 7] = xs[baseE[e]];
        }
        // ---- stage B tile: 128 out-channels x 32 K (vectorized 16B) ----
#pragma unroll
        for (int e = 0; e < 2; ++e) {
            int ci = tid + e * 256;          // 512 chunks of 8 bf16
            int nn = ci >> 2;
            int kc = ci & 3;
            const uint4* src = reinterpret_cast<const uint4*>(
                Wt + (size_t)(n0 + nn) * K_ + k0 + kc * 8);
            *reinterpret_cast<uint4*>(&Bs[nn][kc * 8]) = *src;
        }
        // prefetch next weight tile (global_prefetch_b8 on gfx1250)
        if (k0 + 32 < K_) {
            __builtin_prefetch(Wt + (size_t)(n0 + (tid >> 1)) * K_ + k0 + 32, 0, 0);
        }
        __syncthreads();

        // ---- assemble fragments per ISA layouts & issue WMMA ----
        Frag16 fa[2], fb[4];
#pragma unroll
        for (int mt = 0; mt < 2; ++mt) {
            const unsigned short* row = &As[wm * 32 + mt * 16 + l15][0];
            fa[mt].q[0] = *reinterpret_cast<const uint4*>(row + kA);
            fa[mt].q[1] = *reinterpret_cast<const uint4*>(row + 16 + kA);
        }
#pragma unroll
        for (int nt = 0; nt < 4; ++nt) {
            const unsigned short* row = &Bs[wn * 64 + nt * 16 + l15][0];
            fb[nt].q[0] = *reinterpret_cast<const uint4*>(row + kA);
            fb[nt].q[1] = *reinterpret_cast<const uint4*>(row + 16 + kA);
        }
#pragma unroll
        for (int mt = 0; mt < 2; ++mt)
#pragma unroll
            for (int nt = 0; nt < 4; ++nt)
                acc[mt][nt] = __builtin_amdgcn_wmma_f32_16x16x32_bf16(
                    false, fa[mt].v, false, fb[nt].v,
                    (short)0, acc[mt][nt], false, false);
        __syncthreads();
    }

    // ---- epilogue: bias + ReLU, store bf16 h[m][n] ----
#pragma unroll
    for (int nt = 0; nt < 4; ++nt) {
        int ng = n0 + wn * 64 + nt * 16 + l15;
        float bv = bias[ng];
#pragma unroll
        for (int mt = 0; mt < 2; ++mt) {
#pragma unroll
            for (int v = 0; v < 8; ++v) {
                int mg = m0 + wm * 32 + mt * 16 + v + 8 * half;   // C/D row map
                float val = acc[mt][nt][v] + bv;
                val = fmaxf(val, 0.0f);
                h[(size_t)mg * N_ + ng] = f32_to_bf16(val);
            }
        }
    }
}

// ---------------------------------------------------------------------------
// 4) 1x1 heads: loc (36ch) + score (18ch). One thread per (m, n).
// ---------------------------------------------------------------------------
__global__ void heads_kernel(const unsigned short* __restrict__ h,
                             const float* __restrict__ loc_w,
                             const float* __restrict__ loc_b,
                             const float* __restrict__ score_w,
                             const float* __restrict__ score_b,
                             float* __restrict__ locraw,
                             float* __restrict__ scoreraw) {
    int id = blockIdx.x * blockDim.x + threadIdx.x;
    if (id >= M_ * 54) return;
    int m = id / 54;
    int n = id - m * 54;
    const float* wrow;
    float sum;
    float* dst;
    if (n < 36) { wrow = loc_w + (size_t)n * 512;          sum = loc_b[n];
                  dst  = &locraw[(size_t)m * 36 + n]; }
    else        { int ns = n - 36;
                  wrow = score_w + (size_t)ns * 512;       sum = score_b[ns];
                  dst  = &scoreraw[(size_t)m * 18 + ns]; }
    const uint2*  hp = reinterpret_cast<const uint2*>(h + (size_t)m * 512);
    const float4* wp = reinterpret_cast<const float4*>(wrow);
#pragma unroll 4
    for (int c4 = 0; c4 < 128; ++c4) {
        uint2  hv = hp[c4];
        float4 wv = wp[c4];
        sum += bf16lo_f32(hv.x) * wv.x + bf16hi_f32(hv.x) * wv.y +
               bf16lo_f32(hv.y) * wv.z + bf16hi_f32(hv.y) * wv.w;
    }
    *dst = sum;
}

// ---------------------------------------------------------------------------
// 5) proposal: emit rpn_locs / rpn_scores / anchors, decode+clip+filter,
//    fg softmax score; stash boxes + masked scores for NMS.
// ---------------------------------------------------------------------------
__global__ void proposal_kernel(const float* __restrict__ locraw,
                                const float* __restrict__ scoreraw,
                                const int* __restrict__ img_h_p,
                                const int* __restrict__ img_w_p,
                                const int* __restrict__ scale_p,
                                float* __restrict__ out_locs,
                                float* __restrict__ out_scores,
                                float* __restrict__ out_anchor,
                                float* __restrict__ gboxes,
                                float* __restrict__ gscores) {
    int id = blockIdx.x * blockDim.x + threadIdx.x;
    if (id >= B_ * NANCH) return;
    int b    = id / NANCH;
    int r    = id - b * NANCH;
    int cell = r / A_;
    int a    = r - cell * A_;
    int y    = cell >> 6;
    int x    = cell & 63;
    int m    = (b << 12) + cell;

    float dy = locraw[(size_t)m * 36 + a * 4 + 0];
    float dx = locraw[(size_t)m * 36 + a * 4 + 1];
    float dh = locraw[(size_t)m * 36 + a * 4 + 2];
    float dw = locraw[(size_t)m * 36 + a * 4 + 3];
    float s0 = scoreraw[(size_t)m * 18 + a * 2 + 0];
    float s1 = scoreraw[(size_t)m * 18 + a * 2 + 1];

    size_t ro = (size_t)b * NANCH + r;
    out_locs[ro * 4 + 0] = dy;  out_locs[ro * 4 + 1] = dx;
    out_locs[ro * 4 + 2] = dh;  out_locs[ro * 4 + 3] = dw;
    out_scores[ro * 2 + 0] = s0;
    out_scores[ro * 2 + 1] = s1;

    // anchor geometry
    int ri = a / 3, si = a - ri * 3;
    float ratio = (ri == 0) ? 0.5f : ((ri == 1) ? 1.0f : 2.0f);
    float scl   = (si == 0) ? 8.0f : ((si == 1) ? 16.0f : 32.0f);
    float ah = 16.0f * scl * sqrtf(ratio);
    float aw = 16.0f * scl * sqrtf(1.0f / ratio);
    float acy = y * 16.0f + 8.0f;
    float acx = x * 16.0f + 8.0f;
    if (b == 0) {
        out_anchor[(size_t)r * 4 + 0] = acy - 0.5f * ah;
        out_anchor[(size_t)r * 4 + 1] = acx - 0.5f * aw;
        out_anchor[(size_t)r * 4 + 2] = acy + 0.5f * ah;
        out_anchor[(size_t)r * 4 + 3] = acx + 0.5f * aw;
    }

    // decode + clip + min-size filter
    float cy = dy * ah + acy;
    float cx = dx * aw + acx;
    float bh = __expf(dh) * ah;
    float bw = __expf(dw) * aw;
    float Himg = (float)(*img_h_p);
    float Wimg = (float)(*img_w_p);
    float msz  = 16.0f * (float)(*scale_p);
    float y1 = fminf(fmaxf(cy - 0.5f * bh, 0.0f), Himg);
    float x1 = fminf(fmaxf(cx - 0.5f * bw, 0.0f), Wimg);
    float y2 = fminf(fmaxf(cy + 0.5f * bh, 0.0f), Himg);
    float x2 = fminf(fmaxf(cx + 0.5f * bw, 0.0f), Wimg);
    bool valid = ((y2 - y1) >= msz) && ((x2 - x1) >= msz);

    float fg = 1.0f / (1.0f + __expf(s0 - s1));     // softmax fg channel

    gboxes[(size_t)id * 4 + 0] = y1;
    gboxes[(size_t)id * 4 + 1] = x1;
    gboxes[(size_t)id * 4 + 2] = y2;
    gboxes[(size_t)id * 4 + 3] = x2;
    gscores[id] = valid ? fg : -INFINITY;
}

// ---------------------------------------------------------------------------
// 6) greedy NMS: one 1024-thread workgroup per image; scores live in LDS
//    (147KB -- CDNA5 has 320KB/WGP). Argmax tie-breaks on lowest index to
//    match jnp.argmax. Selection order == top_k order for survivors.
// ---------------------------------------------------------------------------
__global__ __launch_bounds__(1024)
void nms_kernel(const float* __restrict__ gboxes,
                const float* __restrict__ gscores,
                float* __restrict__ out_rois,
                int* __restrict__ out_inds) {
    extern __shared__ float lds[];
    float* sc   = lds;                     // 36864 scores
    float* redS = lds + NANCH;             // 1024
    int*   redI = (int*)(redS + 1024);     // 1024

    const int b = blockIdx.x;
    const int t = threadIdx.x;
    const float* boxes = gboxes + (size_t)b * NANCH * 4;

#pragma unroll
    for (int u = 0; u < 36; ++u)
        sc[t + u * 1024] = gscores[(size_t)b * NANCH + t + u * 1024];
    __syncthreads();

    for (int i = 0; i < NPOST; ++i) {
        // local argmax (first-index tie-break)
        float best = -INFINITY;
        int bidx = 0x7FFFFFFF;
#pragma unroll
        for (int u = 0; u < 36; ++u) {
            int e = t + u * 1024;
            float s = sc[e];
            if (s > best || (s == best && e < bidx)) { best = s; bidx = e; }
        }
        redS[t] = best;
        redI[t] = bidx;
        __syncthreads();
        for (int stride = 512; stride > 0; stride >>= 1) {
            if (t < stride) {
                float s2 = redS[t + stride];
                int   i2 = redI[t + stride];
                if (s2 > redS[t] || (s2 == redS[t] && i2 < redI[t])) {
                    redS[t] = s2; redI[t] = i2;
                }
            }
            __syncthreads();
        }
        int j = redI[0];

        if (t == 0) {
            out_rois[((size_t)b * NPOST + i) * 4 + 0] = boxes[(size_t)j * 4 + 0];
            out_rois[((size_t)b * NPOST + i) * 4 + 1] = boxes[(size_t)j * 4 + 1];
            out_rois[((size_t)b * NPOST + i) * 4 + 2] = boxes[(size_t)j * 4 + 2];
            out_rois[((size_t)b * NPOST + i) * 4 + 3] = boxes[(size_t)j * 4 + 3];
            out_inds[(size_t)b * NPOST + i] = b;
        }

        float by1 = boxes[(size_t)j * 4 + 0];
        float bx1 = boxes[(size_t)j * 4 + 1];
        float by2 = boxes[(size_t)j * 4 + 2];
        float bx2 = boxes[(size_t)j * 4 + 3];
        float areaj = (by2 - by1) * (bx2 - bx1);

#pragma unroll 4
        for (int u = 0; u < 36; ++u) {
            int e = t + u * 1024;
            float s = sc[e];
            if (s == -INFINITY) continue;
            if (e == j) { sc[e] = -INFINITY; continue; }
            float ey1 = boxes[(size_t)e * 4 + 0];
            float ex1 = boxes[(size_t)e * 4 + 1];
            float ey2 = boxes[(size_t)e * 4 + 2];
            float ex2 = boxes[(size_t)e * 4 + 3];
            float yy1 = fmaxf(by1, ey1), xx1 = fmaxf(bx1, ex1);
            float yy2 = fminf(by2, ey2), xx2 = fminf(bx2, ex2);
            float inter = fmaxf(yy2 - yy1, 0.0f) * fmaxf(xx2 - xx1, 0.0f);
            float areae = (ey2 - ey1) * (ex2 - ex1);
            float iou = inter / (areaj + areae - inter);
            if (iou > 0.7f) sc[e] = -INFINITY;
        }
        __syncthreads();   // scores stable + j consumed before next iteration
    }
}

// ---------------------------------------------------------------------------
extern "C" void kernel_launch(void* const* d_in, const int* in_sizes, int n_in,
                              void* d_out, int out_size, void* d_ws, size_t ws_size,
                              hipStream_t stream) {
    (void)in_sizes; (void)n_in; (void)out_size; (void)ws_size;

    const float* x        = (const float*)d_in[0];
    const float* conv1_w  = (const float*)d_in[1];
    const float* conv1_b  = (const float*)d_in[2];
    const float* score_w  = (const float*)d_in[3];
    const float* score_b  = (const float*)d_in[4];
    const float* loc_w    = (const float*)d_in[5];
    const float* loc_b    = (const float*)d_in[6];
    const int*   img_h    = (const int*)d_in[7];
    const int*   img_w    = (const int*)d_in[8];
    const int*   scale    = (const int*)d_in[9];

    char* ws = (char*)d_ws;
    unsigned short* xp       = (unsigned short*)(ws + OFF_XP);
    unsigned short* Wt       = (unsigned short*)(ws + OFF_WT);
    unsigned short* hbuf     = (unsigned short*)(ws + OFF_H);
    float*          locraw   = (float*)(ws + OFF_LR);
    float*          scoreraw = (float*)(ws + OFF_SR);
    float*          gboxes   = (float*)(ws + OFF_BX);
    float*          gscores  = (float*)(ws + OFF_SC);

    float* out_f      = (float*)d_out;
    float* out_locs   = out_f + OUT_LOCS;
    float* out_scores = out_f + OUT_SCORES;
    float* out_rois   = out_f + OUT_ROIS;
    int*   out_inds   = (int*)(out_f + OUT_INDS);
    float* out_anchor = out_f + OUT_ANCHOR;

    // 1) pad+convert activations
    {
        size_t total = (size_t)B_ * C_ * HP2;
        int blocks = (int)((total + 255) / 256);
        prep_x_kernel<<<blocks, 256, 0, stream>>>(x, xp);
    }
    // 2) convert+permute weights
    {
        size_t total = (size_t)N_ * K_;
        int blocks = (int)((total + 255) / 256);
        prep_w_kernel<<<blocks, 256, 0, stream>>>(conv1_w, Wt);
    }
    // 3) WMMA implicit-GEMM conv + bias + ReLU
    {
        dim3 grid(M_ / 128, N_ / 128);
        conv3x3_wmma_kernel<<<grid, 256, 0, stream>>>(xp, Wt, conv1_b, hbuf);
    }
    // 4) 1x1 heads
    {
        int total = M_ * 54;
        heads_kernel<<<(total + 255) / 256, 256, 0, stream>>>(
            hbuf, loc_w, loc_b, score_w, score_b, locraw, scoreraw);
    }
    // 5) proposals + anchors + raw outputs
    {
        int total = B_ * NANCH;
        proposal_kernel<<<(total + 255) / 256, 256, 0, stream>>>(
            locraw, scoreraw, img_h, img_w, scale,
            out_locs, out_scores, out_anchor, gboxes, gscores);
    }
    // 6) NMS: one block per image, scores in LDS
    {
        size_t shmem = (size_t)(NANCH + 1024) * sizeof(float) + 1024 * sizeof(int);
        nms_kernel<<<B_, 1024, shmem, stream>>>(gboxes, gscores, out_rois, out_inds);
    }
}
